// GraphAttentionLayer_25520695673167
// MI455X (gfx1250) — compile-verified, matrix operand order checked by reading, not run on hardware
//
#include <hip/hip_runtime.h>
#include <hip/hip_bf16.h>

typedef __attribute__((ext_vector_type(16))) _Float16 v16h;
typedef __attribute__((ext_vector_type(8)))  _Float16 v8h;
typedef __attribute__((ext_vector_type(4)))  _Float16 v4h;
typedef __attribute__((ext_vector_type(8)))  float    v8f;

#define N_NODES 8192
#define F_IN    512
#define F_OUT   64
#define J_SPLIT 4
#define J_CHUNK (N_NODES / J_SPLIT)   // 2048
#define GAT_ALPHA 0.2f
#define ATTN_WAVES 4

union V16U { v16h v; v8h h[2]; v4h q[4]; };

// ---- helpers ---------------------------------------------------------------

__device__ __forceinline__ v4h gat_cvt4(float4 x) {
  v4h r;
  r[0] = (_Float16)x.x; r[1] = (_Float16)x.y;
  r[2] = (_Float16)x.z; r[3] = (_Float16)x.w;
  return r;
}

// e = leakyrelu(wh1 + wh2), s = adj>0 ? exp(e) : 0 ; accumulate row sum.
__device__ __forceinline__ v4h gat_edge4(int4 ad, float4 w2, float wh1, float& lsum) {
  v4h r; float e, s;
  e = wh1 + w2.x; e = e > 0.f ? e : GAT_ALPHA * e; s = (ad.x > 0) ? __expf(e) : 0.f; lsum += s; r[0] = (_Float16)s;
  e = wh1 + w2.y; e = e > 0.f ? e : GAT_ALPHA * e; s = (ad.y > 0) ? __expf(e) : 0.f; lsum += s; r[1] = (_Float16)s;
  e = wh1 + w2.z; e = e > 0.f ? e : GAT_ALPHA * e; s = (ad.z > 0) ? __expf(e) : 0.f; lsum += s; r[2] = (_Float16)s;
  e = wh1 + w2.w; e = e > 0.f ? e : GAT_ALPHA * e; s = (ad.w > 0) ? __expf(e) : 0.f; lsum += s; r[3] = (_Float16)s;
  return r;
}

// async global->LDS copy of 16 bytes per lane (tracked by ASYNCcnt)
__device__ __forceinline__ void gat_async_b128(unsigned lds_byte_off, const void* g) {
  asm volatile("global_load_async_to_lds_b128 %0, %1, off"
               :: "v"(lds_byte_off), "v"(g) : "memory");
}

// ---- kernel 1: W [512,64] f32 -> WT [64,512] f16 ---------------------------

__global__ void gat_wt(const float* __restrict__ W, _Float16* __restrict__ WT) {
  int idx = blockIdx.x * blockDim.x + threadIdx.x;   // < 32768
  int k = idx >> 6, n = idx & 63;
  WT[(size_t)n * F_IN + k] = (_Float16)W[idx];
}

// ---- kernel 2: Wh = h @ W via WMMA; also emit WhT (f16, transposed) --------

__global__ void gat_gemm1(const float* __restrict__ h, const _Float16* __restrict__ WT,
                          float* __restrict__ Wh, _Float16* __restrict__ WhT) {
  const int lane = threadIdx.x & 31;
  const int ib   = blockIdx.x * (blockDim.x >> 5) + (threadIdx.x >> 5);  // 0..511
  const int i0   = ib * 16;
  const int m    = lane & 15;
  const int hi   = lane >> 4;
  const int kb   = hi * 8;         // A-operand K base per ISA VGPR layout

  v8f zero = {};
  v8f acc[4] = {zero, zero, zero, zero};
  const float* hrow = h + (size_t)(i0 + m) * F_IN;

  for (int k0 = 0; k0 < F_IN; k0 += 32) {
    V16U A;
    float4 a0 = *(const float4*)(hrow + k0 + kb);
    float4 a1 = *(const float4*)(hrow + k0 + kb + 4);
    float4 a2 = *(const float4*)(hrow + k0 + 16 + kb);
    float4 a3 = *(const float4*)(hrow + k0 + 16 + kb + 4);
    A.q[0] = gat_cvt4(a0); A.q[1] = gat_cvt4(a1);
    A.q[2] = gat_cvt4(a2); A.q[3] = gat_cvt4(a3);

#pragma unroll
    for (int t = 0; t < 4; ++t) {
      V16U B;   // lane holds column n = m, K set mirrors A layout
      const _Float16* bp = WT + (size_t)(16 * t + m) * F_IN + k0 + kb;
      B.h[0] = *(const v8h*)(bp);
      B.h[1] = *(const v8h*)(bp + 16);
      acc[t] = __builtin_amdgcn_wmma_f32_16x16x32_f16(false, A.v, false, B.v,
                                                      (short)0, acc[t], false, false);
    }
  }

  // C/D layout: VGPR r, lane L -> (M = r + 8*(L>>4), N = L&15)
#pragma unroll
  for (int t = 0; t < 4; ++t) {
#pragma unroll
    for (int r = 0; r < 8; ++r) {
      int row = i0 + r + 8 * hi;
      int col = 16 * t + m;
      float val = acc[t][r];
      Wh [(size_t)row * F_OUT + col]   = val;
      WhT[(size_t)col * N_NODES + row] = (_Float16)val;
    }
  }
}

// ---- kernel 3: Wh1 = Wh@a[:64], Wh2 = Wh@a[64:] ----------------------------

__global__ void gat_rowdot(const float* __restrict__ Wh, const float* __restrict__ a,
                           float* __restrict__ Wh1, float* __restrict__ Wh2) {
  int i = blockIdx.x * blockDim.x + threadIdx.x;     // < 8192
  const float4* row = (const float4*)(Wh + (size_t)i * F_OUT);
  const float4* a1  = (const float4*)a;
  const float4* a2  = (const float4*)(a + F_OUT);
  float s1 = 0.f, s2 = 0.f;
#pragma unroll
  for (int q = 0; q < F_OUT / 4; ++q) {
    float4 v = row[q], x = a1[q], y = a2[q];
    s1 += v.x * x.x + v.y * x.y + v.z * x.z + v.w * x.w;
    s2 += v.x * y.x + v.y * y.y + v.z * y.z + v.w * y.w;
  }
  Wh1[i] = s1; Wh2[i] = s2;
}

// ---- kernel 4: fused masked-softmax-numerator + (s @ Wh) via WMMA ----------
// All 4 waves of a block share one j-split -> identical B tiles, staged once
// per j-tile into LDS with double-buffered GLOBAL_LOAD_ASYNC_TO_LDS_B128.

__global__ void gat_attn(const int* __restrict__ adj, const _Float16* __restrict__ WhT,
                         const float* __restrict__ Wh1, const float* __restrict__ Wh2,
                         float* __restrict__ accP, float* __restrict__ lpart) {
  __shared__ __align__(16) _Float16 ldsB[2][64 * 32];   // [buf][n*32 + k], 4KB each

  const int tid  = threadIdx.x;
  const int lane = tid & 31;
  const int wv   = tid >> 5;                               // 0..3
  const int js   = blockIdx.x >> 7;                        // 0..3
  const int ib   = (blockIdx.x & 127) * ATTN_WAVES + wv;   // 0..511
  const int i0   = ib * 16;
  const int m    = lane & 15;
  const int hi   = lane >> 4;
  const int kb   = hi * 8;
  const int jbeg = js * J_CHUNK;
  const int jend = jbeg + J_CHUNK;

  // cooperative B staging: 4KB tile = 256 x 16B chunks; thread t owns chunks
  // (n0 = t>>2, part = t&3) and (n0+32, part)
  const int n0   = tid >> 2;
  const int part = tid & 3;

  const float wh1m = Wh1[i0 + m];
  const int* arow  = adj + (size_t)(i0 + m) * N_NODES;

  float lsum = 0.f;
  v8f zero = {};
  v8f acc[4] = {zero, zero, zero, zero};

  auto stage = [&](int buf, int j0) {
    unsigned base = (unsigned)(uintptr_t)(&ldsB[buf][0]);
    gat_async_b128(base + (unsigned)(n0 * 64 + part * 16),
                   WhT + (size_t)n0 * N_NODES + j0 + part * 8);
    gat_async_b128(base + (unsigned)((n0 + 32) * 64 + part * 16),
                   WhT + (size_t)(n0 + 32) * N_NODES + j0 + part * 8);
  };

  stage(0, jbeg);
  int cur = 0;

  for (int j0 = jbeg; j0 < jend; j0 += 32) {
    // prefetch next tile into the other buffer (wrap keeps the wait-imm constant)
    int jn = (j0 + 32 < jend) ? (j0 + 32) : jbeg;
    stage(cur ^ 1, jn);

    // current tile's 2 async wave-ops are the oldest outstanding -> wait <= 2
    asm volatile("s_wait_asynccnt 0x2" ::: "memory");
    __syncthreads();

    // pull the adj stream ahead (16 cachelines needed 16 tiles from now)
    __builtin_prefetch(arow + ((j0 + 512 < jend) ? j0 + 512 : jbeg), 0, 3);

    // adjacency tile: lane covers row m, K columns in WMMA-A register order
    int4 d0 = *(const int4*)(arow + j0 + kb);
    int4 d1 = *(const int4*)(arow + j0 + kb + 4);
    int4 d2 = *(const int4*)(arow + j0 + 16 + kb);
    int4 d3 = *(const int4*)(arow + j0 + 16 + kb + 4);
    float4 w0 = *(const float4*)(Wh2 + j0 + kb);
    float4 w1 = *(const float4*)(Wh2 + j0 + kb + 4);
    float4 w2 = *(const float4*)(Wh2 + j0 + 16 + kb);
    float4 w3 = *(const float4*)(Wh2 + j0 + 16 + kb + 4);

    V16U P;   // unnormalized attention weights, packed directly in A layout
    P.q[0] = gat_edge4(d0, w0, wh1m, lsum);
    P.q[1] = gat_edge4(d1, w1, wh1m, lsum);
    P.q[2] = gat_edge4(d2, w2, wh1m, lsum);
    P.q[3] = gat_edge4(d3, w3, wh1m, lsum);

#pragma unroll
    for (int t = 0; t < 4; ++t) {
      V16U B;   // B from LDS: lane holds output feature n = m
      const _Float16* bp = &ldsB[cur][(16 * t + m) * 32 + kb];
      B.h[0] = *(const v8h*)(bp);
      B.h[1] = *(const v8h*)(bp + 16);
      acc[t] = __builtin_amdgcn_wmma_f32_16x16x32_f16(false, P.v, false, B.v,
                                                      (short)0, acc[t], false, false);
    }

    __syncthreads();   // all waves done reading buf[cur] before it is refilled
    cur ^= 1;
  }

  // lanes m and m+16 both hold partials of row m's denominator
  float ltot = lsum + __shfl_xor(lsum, 16, 32);
  if (hi == 0) lpart[js * N_NODES + i0 + m] = ltot;

  float* obase = accP + (size_t)js * N_NODES * F_OUT;
#pragma unroll
  for (int t = 0; t < 4; ++t) {
#pragma unroll
    for (int r = 0; r < 8; ++r) {
      obase[(size_t)(i0 + r + 8 * hi) * F_OUT + 16 * t + m] = acc[t][r];
    }
  }
}

// ---- kernel 5: combine j-splits, divide by softmax denom, ELU --------------

__global__ void gat_combine(const float* __restrict__ accP, const float* __restrict__ lpart,
                            float* __restrict__ out) {
  int idx = blockIdx.x * blockDim.x + threadIdx.x;   // < 8192*64
  int i = idx >> 6;
  const int NF = N_NODES * F_OUT;
  float s = accP[idx] + accP[idx + NF] + accP[idx + 2 * NF] + accP[idx + 3 * NF];
  float l = lpart[i] + lpart[i + N_NODES] + lpart[i + 2 * N_NODES] + lpart[i + 3 * N_NODES];
  float v = s / l;
  out[idx] = v > 0.f ? v : (__expf(v) - 1.f);
}

// ---- launch ----------------------------------------------------------------

extern "C" void kernel_launch(void* const* d_in, const int* in_sizes, int n_in,
                              void* d_out, int out_size, void* d_ws, size_t ws_size,
                              hipStream_t stream) {
  const float* h   = (const float*)d_in[0];
  const int*   adj = (const int*)  d_in[1];
  const float* W   = (const float*)d_in[2];
  const float* a   = (const float*)d_in[3];
  float* out = (float*)d_out;

  char* ws = (char*)d_ws;
  _Float16* WT    = (_Float16*)(ws + 0);         //  64KB: W^T f16 [64,512]
  _Float16* WhT   = (_Float16*)(ws + 65536);     //   1MB: Wh^T f16 [64,8192]
  float*    Wh    = (float*)   (ws + 1114112);   //   2MB: Wh f32 [8192,64]
  float*    Wh1   = (float*)   (ws + 3211264);   //  32KB
  float*    Wh2   = (float*)   (ws + 3244032);   //  32KB
  float*    lpart = (float*)   (ws + 3276800);   // 128KB: [4,8192]
  float*    accP  = (float*)   (ws + 3407872);   //   8MB: [4,8192,64]

  gat_wt     <<<128, 256, 0, stream>>>(W, WT);
  gat_gemm1  <<<128, 128, 0, stream>>>(h, WT, Wh, WhT);
  gat_rowdot <<< 32, 256, 0, stream>>>(Wh, a, Wh1, Wh2);
  gat_attn   <<<512, 128, 0, stream>>>(adj, WhT, Wh1, Wh2, accP, lpart);
  gat_combine<<<2048, 256, 0, stream>>>(accP, lpart, out);
}
